// MoeFFN_51273319580267
// MI455X (gfx1250) — matrix operand order repaired and checked by reading, not compiled, not run on hardware
//
#include <hip/hip_runtime.h>
#include <hip/hip_bf16.h>
#include <math.h>

// ---------------------------------------------------------------------------
// MoE FFN (top-2, 8 experts) for gfx1250: bf16 WMMA grouped GEMM pipeline.
// GEMM1: LDS-staged (async copy) gathered A, 32x32 per-wave tiles, up+gate fused
// GEMM2: direct-global fragments, 32x64 per-wave tiles, fused weighted scatter
// ---------------------------------------------------------------------------

typedef __attribute__((ext_vector_type(16))) __bf16 v16bf;
typedef __attribute__((ext_vector_type(8)))  float  v8f;

constexpr int SEQ     = 8192;
constexpr int IN_DIM  = 1024;
constexpr int HID     = 2048;
constexpr int NEXP    = 8;
constexpr int TOPK    = 2;
constexpr int NTOK    = SEQ * TOPK;          // 16384 dispatched tokens
constexpr int MALIGN  = 64;                  // bucket row alignment = block M tile
constexpr int CAP     = NTOK + NEXP * MALIGN;   // 16896 padded rows
constexpr int M64     = CAP / MALIGN;        // 264 M-tiles of 64 rows

union ABFrag { v16bf v; uint4 u4[2]; };

#if defined(__gfx1250__) && __has_builtin(__builtin_amdgcn_global_load_async_to_lds_b128)
#define HAVE_ASYNC_LDS 1
#endif

#ifdef HAVE_ASYNC_LDS
typedef int v4i_vec __attribute__((vector_size(16)));
typedef __attribute__((address_space(1))) v4i_vec gv4i;
typedef __attribute__((address_space(3))) v4i_vec lv4i;
#endif

__device__ __forceinline__ void copy16_async(const unsigned short* gsrc, unsigned short* ldst) {
#ifdef HAVE_ASYNC_LDS
    __builtin_amdgcn_global_load_async_to_lds_b128((gv4i*)gsrc, (lv4i*)ldst, 0, 0);
#else
    *(uint4*)ldst = *(const uint4*)gsrc;
#endif
}

__device__ __forceinline__ void wait_async() {
#ifdef HAVE_ASYNC_LDS
#if __has_builtin(__builtin_amdgcn_s_wait_asynccnt)
    __builtin_amdgcn_s_wait_asynccnt(0);
#else
    asm volatile("s_wait_asynccnt 0" ::: "memory");
#endif
#endif
}

__device__ __forceinline__ unsigned short f2bf(float f) {
    unsigned int u = __builtin_bit_cast(unsigned int, f);
    u += 0x7FFFu + ((u >> 16) & 1u);         // round-to-nearest-even
    return (unsigned short)(u >> 16);
}

// gelu(tanh approx) = g * sigmoid(1.5957691*(g + 0.044715 g^3)); branchless
__device__ __forceinline__ float gelu_tanh(float g) {
    float u = 1.5957691216057308f * (g + 0.044715f * g * g * g);
    return g / (1.0f + __expf(-u));
}

// ---------------------------------------------------------------------------
// Bucketing kernels
// meta[0..8]: aligned expert offsets  meta[16..23]: counts  meta[24..31]: cursors
// ---------------------------------------------------------------------------
__global__ void k_init_meta(int* meta) {
    int i = threadIdx.x;
    if (i < 16) meta[16 + i] = 0;
}

__global__ void k_fill_tok(int* tok) {
    int i = blockIdx.x * blockDim.x + threadIdx.x;
    if (i < CAP) tok[i] = -1;
}

__global__ void k_count(const int* __restrict__ sel, int* meta) {
    int t = blockIdx.x * blockDim.x + threadIdx.x;
    if (t < NTOK) atomicAdd(&meta[16 + (sel[t] & 7)], 1);
}

__global__ void k_scan(int* meta) {
    if (threadIdx.x == 0 && blockIdx.x == 0) {
        int acc = 0;
        meta[0] = 0;
        #pragma unroll
        for (int e = 0; e < NEXP; ++e) {
            acc += (meta[16 + e] + (MALIGN - 1)) & ~(MALIGN - 1);
            meta[e + 1] = acc;
        }
    }
}

__global__ void k_scatter(const int* __restrict__ sel, int* meta, int* tok) {
    int t = blockIdx.x * blockDim.x + threadIdx.x;
    if (t < NTOK) {
        int e   = sel[t] & 7;
        int pos = meta[e] + atomicAdd(&meta[24 + e], 1);
        tok[pos] = t;
    }
}

// ---------------------------------------------------------------------------
// fp32 -> bf16 conversion kernels
// ---------------------------------------------------------------------------
__global__ void k_f2bf4(const float* __restrict__ in, unsigned short* __restrict__ out, int n4) {
    int i = blockIdx.x * blockDim.x + threadIdx.x;
    if (i < n4) {
        float4 f = ((const float4*)in)[i];
        ushort4 o;
        o.x = f2bf(f.x); o.y = f2bf(f.y); o.z = f2bf(f.z); o.w = f2bf(f.w);
        ((ushort4*)out)[i] = o;
    }
}

// WdT[e][d][h] = bf16(down[e][h][d])
__global__ void k_downT(const float* __restrict__ down, unsigned short* __restrict__ wdt) {
    long long i = (long long)blockIdx.x * blockDim.x + threadIdx.x;
    const long long n = (long long)NEXP * HID * IN_DIM;
    if (i < n) {
        int e = (int)(i / ((long long)HID * IN_DIM));
        int r = (int)(i % ((long long)HID * IN_DIM));
        int h = r / IN_DIM;
        int d = r % IN_DIM;
        wdt[((size_t)e * IN_DIM + d) * HID + h] = f2bf(down[i]);
    }
}

__global__ void k_zero_out(float* __restrict__ out, int n4) {
    int i = blockIdx.x * blockDim.x + threadIdx.x;
    if (i < n4) {
        float4 z = {0.f, 0.f, 0.f, 0.f};
        ((float4*)out)[i] = z;
    }
}

// ---------------------------------------------------------------------------
// GEMM 1: act = gelu(x Wg^T) * (x Wu^T), grouped by expert.
// Block 256 = 8 waves (2 Mx4 N). Block tile 64 x 128. Wave tile 32 x 32 for
// BOTH up and gate (8 accumulator tiles). A staged in LDS via async copy,
// double-buffered 64x64 bf16 chunks; B fragments contiguous from global.
// ---------------------------------------------------------------------------
__global__ __launch_bounds__(256) void k_upgate(
    const unsigned short* __restrict__ Xb,
    const unsigned short* __restrict__ Wub,
    const unsigned short* __restrict__ Wgb,
    const int* __restrict__ meta,
    const int* __restrict__ tok,
    unsigned short* __restrict__ act)
{
    __shared__ unsigned short As[2][64 * 64];
    __shared__ int smTok[64];

    const int tid  = threadIdx.x;
    const int w    = tid >> 5;
    const int lane = tid & 31;
    const int m    = lane & 15;
    const int hi   = lane >> 4;
    const int wm   = w >> 2;          // 0..1  (M row of waves)
    const int wn   = w & 3;           // 0..3  (N col of waves)

    const int p0 = blockIdx.y * 64;
    const int n0 = blockIdx.x * 128 + wn * 32;

    int e = 0;
    #pragma unroll
    for (int i = 0; i < 7; ++i) e += (p0 >= meta[i + 1]) ? 1 : 0;

    if (tid < 64) smTok[tid] = tok[p0 + tid];
    __syncthreads();

    // B base rows: W row (= B column) per N-subtile, contiguous in K
    const unsigned short* urow[2];
    const unsigned short* grow[2];
    #pragma unroll
    for (int nj = 0; nj < 2; ++nj) {
        size_t rb = ((size_t)e * HID + (n0 + nj * 16 + m)) * IN_DIM;
        urow[nj] = Wub + rb;
        grow[nj] = Wgb + rb;
    }

    v8f cu[2][2], cg[2][2];
    #pragma unroll
    for (int mi = 0; mi < 2; ++mi)
        #pragma unroll
        for (int nj = 0; nj < 2; ++nj) {
            cu[mi][nj] = (v8f){0.f,0.f,0.f,0.f,0.f,0.f,0.f,0.f};
            cg[mi][nj] = (v8f){0.f,0.f,0.f,0.f,0.f,0.f,0.f,0.f};
        }

    // stage a 64-row x 64-col bf16 chunk of gathered token rows into LDS
    auto stage = [&](int buf, int kb) {
        #pragma unroll
        for (int u = tid; u < 512; u += 256) {
            int r  = u >> 3;            // 0..63 row within tile
            int c8 = (u & 7) * 8;       // 0..56 col (8 bf16 = 16B units)
            int t  = smTok[r];
            int s  = (t >= 0) ? (t >> 1) : 0;
            copy16_async(Xb + (size_t)s * IN_DIM + kb + c8, &As[buf][r * 64 + c8]);
        }
    };

    stage(0, 0);
    int buf = 0;
    for (int kb = 0; kb < IN_DIM; kb += 64, buf ^= 1) {
        wait_async();
        __syncthreads();
        if (kb + 64 < IN_DIM) stage(buf ^ 1, kb + 64);

        #pragma unroll
        for (int kk = 0; kk < 2; ++kk) {
            const int kc = kk * 32;          // k offset within LDS chunk
            const int kg = kb + kc;          // global k for B

            ABFrag a[2], bu[2], bg[2];
            #pragma unroll
            for (int mi = 0; mi < 2; ++mi) {
                const unsigned short* ar = &As[buf][(wm * 32 + mi * 16 + m) * 64];
                a[mi].u4[0] = *(const uint4*)(ar + kc + hi * 16);
                a[mi].u4[1] = *(const uint4*)(ar + kc + hi * 16 + 8);
            }
            #pragma unroll
            for (int nj = 0; nj < 2; ++nj) {
                bu[nj].u4[0] = *(const uint4*)(urow[nj] + kg + hi * 16);
                bu[nj].u4[1] = *(const uint4*)(urow[nj] + kg + hi * 16 + 8);
                bg[nj].u4[0] = *(const uint4*)(grow[nj] + kg + hi * 16);
                bg[nj].u4[1] = *(const uint4*)(grow[nj] + kg + hi * 16 + 8);
            }
            #pragma unroll
            for (int mi = 0; mi < 2; ++mi)
                #pragma unroll
                for (int nj = 0; nj < 2; ++nj) {
                    cu[mi][nj] = __builtin_amdgcn_wmma_f32_16x16x32_bf16(
                        false, a[mi].v, false, bu[nj].v, (short)0, cu[mi][nj], false, false);
                    cg[mi][nj] = __builtin_amdgcn_wmma_f32_16x16x32_bf16(
                        false, a[mi].v, false, bg[nj].v, (short)0, cg[mi][nj], false, false);
                }
        }
        __syncthreads();
    }

    // epilogue: act = gelu(gate) * up  (C layout: row = hi*8+r, col = lane&15)
    #pragma unroll
    for (int mi = 0; mi < 2; ++mi)
        #pragma unroll
        for (int nj = 0; nj < 2; ++nj)
            #pragma unroll
            for (int r = 0; r < 8; ++r) {
                int row = p0 + wm * 32 + mi * 16 + hi * 8 + r;
                int col = n0 + nj * 16 + m;
                float hv = cu[mi][nj][r];
                float gv = cg[mi][nj][r];
                act[(size_t)row * HID + col] = f2bf(gelu_tanh(gv) * hv);
            }
}

// ---------------------------------------------------------------------------
// GEMM 2: y = act @ W_down[e]; out[token] += weight * y (atomic combine).
// Block 256 = 8 waves (2 M x 4 N). Block tile 64 x 256. Wave tile 32 x 64.
// A rows contiguous (sorted space), B rows contiguous (pre-transposed WdT).
// ---------------------------------------------------------------------------
__global__ __launch_bounds__(256) void k_down(
    const unsigned short* __restrict__ act,
    const unsigned short* __restrict__ WdT,
    const int* __restrict__ meta,
    const int* __restrict__ tok,
    const float* __restrict__ wts,
    float* __restrict__ out)
{
    __shared__ int smTok[64];

    const int tid  = threadIdx.x;
    const int w    = tid >> 5;
    const int lane = tid & 31;
    const int m    = lane & 15;
    const int hi   = lane >> 4;
    const int wm   = w >> 2;          // 0..1
    const int wn   = w & 3;           // 0..3

    const int p0 = blockIdx.y * 64;
    const int n0 = blockIdx.x * 256 + wn * 64;

    int e = 0;
    #pragma unroll
    for (int i = 0; i < 7; ++i) e += (p0 >= meta[i + 1]) ? 1 : 0;

    if (tid < 64) smTok[tid] = tok[p0 + tid];
    __syncthreads();

    const unsigned short* arow[2];
    #pragma unroll
    for (int mi = 0; mi < 2; ++mi)
        arow[mi] = act + (size_t)(p0 + wm * 32 + mi * 16 + m) * HID;

    const unsigned short* brow[4];
    #pragma unroll
    for (int nj = 0; nj < 4; ++nj)
        brow[nj] = WdT + ((size_t)e * IN_DIM + (n0 + nj * 16 + m)) * HID;

    v8f c[2][4];
    #pragma unroll
    for (int mi = 0; mi < 2; ++mi)
        #pragma unroll
        for (int nj = 0; nj < 4; ++nj)
            c[mi][nj] = (v8f){0.f,0.f,0.f,0.f,0.f,0.f,0.f,0.f};

    #pragma unroll 2
    for (int k0 = 0; k0 < HID; k0 += 32) {
        ABFrag a[2], b[4];
        #pragma unroll
        for (int mi = 0; mi < 2; ++mi) {
            a[mi].u4[0] = *(const uint4*)(arow[mi] + k0 + hi * 16);
            a[mi].u4[1] = *(const uint4*)(arow[mi] + k0 + hi * 16 + 8);
        }
        #pragma unroll
        for (int nj = 0; nj < 4; ++nj) {
            b[nj].u4[0] = *(const uint4*)(brow[nj] + k0 + hi * 16);
            b[nj].u4[1] = *(const uint4*)(brow[nj] + k0 + hi * 16 + 8);
        }
        #pragma unroll
        for (int mi = 0; mi < 2; ++mi)
            #pragma unroll
            for (int nj = 0; nj < 4; ++nj)
                c[mi][nj] = __builtin_amdgcn_wmma_f32_16x16x32_bf16(
                    false, a[mi].v, false, b[nj].v, (short)0, c[mi][nj], false, false);
    }

    // weighted scatter-combine into d_out
    #pragma unroll
    for (int mi = 0; mi < 2; ++mi)
        #pragma unroll
        for (int r = 0; r < 8; ++r) {
            int lrow = wm * 32 + mi * 16 + hi * 8 + r;
            int t    = smTok[lrow];
            if (t >= 0) {
                float wgt = wts[t];                  // weights is (SEQ, TOPK) flat
                float* orow = out + (size_t)(t >> 1) * IN_DIM;
                #pragma unroll
                for (int nj = 0; nj < 4; ++nj)
                    unsafeAtomicAdd(orow + n0 + nj * 16 + m, wgt * c[mi][nj][r]);
            }
        }
}

// ---------------------------------------------------------------------------
// Launch
// ---------------------------------------------------------------------------
extern "C" void kernel_launch(void* const* d_in, const int* in_sizes, int n_in,
                              void* d_out, int out_size, void* d_ws, size_t ws_size,
                              hipStream_t stream) {
    const float* inp   = (const float*)d_in[0];
    const float* wts   = (const float*)d_in[1];
    const float* up    = (const float*)d_in[2];
    const float* gate  = (const float*)d_in[3];
    const float* down  = (const float*)d_in[4];
    const int*   sel   = (const int*)d_in[5];
    float*       out   = (float*)d_out;

    // workspace carve-up
    char* w = (char*)d_ws;
    int*  meta = (int*)w;
    int*  tok  = (int*)(w + 256);
    size_t off = 256 + sizeof(int) * (size_t)CAP;
    off = (off + 255) & ~(size_t)255;
    unsigned short* Xb  = (unsigned short*)(w + off); off += (size_t)SEQ  * IN_DIM * 2;
    unsigned short* Wub = (unsigned short*)(w + off); off += (size_t)NEXP * HID * IN_DIM * 2;
    unsigned short* Wgb = (unsigned short*)(w + off); off += (size_t)NEXP * HID * IN_DIM * 2;
    unsigned short* WdT = (unsigned short*)(w + off); off += (size_t)NEXP * IN_DIM * HID * 2;
    unsigned short* act = (unsigned short*)(w + off); off += (size_t)CAP  * HID * 2;

    // 1) bucket tokens by expert (64-row aligned buckets, -1 sentinels)
    k_init_meta<<<1, 32, 0, stream>>>(meta);
    k_fill_tok<<<(CAP + 255) / 256, 256, 0, stream>>>(tok);
    k_count<<<(NTOK + 255) / 256, 256, 0, stream>>>(sel, meta);
    k_scan<<<1, 32, 0, stream>>>(meta);
    k_scatter<<<(NTOK + 255) / 256, 256, 0, stream>>>(sel, meta, tok);

    // 2) fp32 -> bf16 conversions
    {
        int n4 = SEQ * IN_DIM / 4;
        k_f2bf4<<<(n4 + 255) / 256, 256, 0, stream>>>(inp, Xb, n4);
    }
    {
        int n4 = NEXP * HID * IN_DIM / 4;
        k_f2bf4<<<(n4 + 255) / 256, 256, 0, stream>>>(up,   Wub, n4);
        k_f2bf4<<<(n4 + 255) / 256, 256, 0, stream>>>(gate, Wgb, n4);
    }
    {
        long long n = (long long)NEXP * HID * IN_DIM;
        k_downT<<<(int)((n + 255) / 256), 256, 0, stream>>>(down, WdT);
    }

    // 3) zero output (atomics accumulate into it; must be per-call)
    {
        int n4 = SEQ * IN_DIM / 4;
        k_zero_out<<<(n4 + 255) / 256, 256, 0, stream>>>(out, n4);
    }

    // 4) grouped GEMMs
    dim3 g1(HID / 128, M64);
    k_upgate<<<g1, 256, 0, stream>>>(Xb, Wub, Wgb, meta, tok, act);

    dim3 g2(IN_DIM / 256, M64);
    k_down<<<g2, 256, 0, stream>>>(act, WdT, meta, tok, wts, out);
}
